// ActorCritic_64330020159639
// MI455X (gfx1250) — matrix operand order repaired and checked by reading, not compiled, not run on hardware
//
#include <hip/hip_runtime.h>

#define B_ROWS   65536
#define FEAT     261
#define HID      128
#define ROWS_PB  128              // rows per block
#define NBLK     (B_ROWS / ROWS_PB)
#define NC       9                // K chunks of 32 (288 = 261 padded)

typedef __attribute__((ext_vector_type(16))) __bf16 v16bf;
typedef __attribute__((ext_vector_type(8)))  float  v8f;

#if defined(__has_builtin)
#if __has_builtin(__builtin_amdgcn_cvt_pk_bf16_f32)
#define HAVE_CVT_PK_BF16 1
#endif
#endif

__device__ __forceinline__ unsigned short f2bf(float f) {
    unsigned u = __float_as_uint(f);
    unsigned r = u + 0x7FFFu + ((u >> 16) & 1u);   // round-to-nearest-even
    return (unsigned short)(r >> 16);
}
__device__ __forceinline__ unsigned pack2(float a, float b) {
#ifdef HAVE_CVT_PK_BF16
    typedef __attribute__((ext_vector_type(2))) __bf16 v2bf;
    v2bf p = __builtin_amdgcn_cvt_pk_bf16_f32(a, b);
    return __builtin_bit_cast(unsigned, p);
#else
    return (unsigned)f2bf(a) | ((unsigned)f2bf(b) << 16);
#endif
}
__device__ __forceinline__ float wred32(float s) {
    s += __shfl_xor(s, 1);
    s += __shfl_xor(s, 2);
    s += __shfl_xor(s, 4);
    s += __shfl_xor(s, 8);
    s += __shfl_xor(s, 16);
    return s;
}
// Butterfly halving step: one shuffle advances TWO rows' reductions.
// After it, lanes with bit m == 0 hold u's pair-sum, bit m == 1 hold w's.
__device__ __forceinline__ float halve(float u, float w, int m, int lane) {
    bool hi = (lane & m) != 0;
    float t = hi ? u : w;              // send the one the partner keeps
    float r = __shfl_xor(t, m);
    return (hi ? w : u) + r;
}
__device__ __forceinline__ int swz(int idx) {     // LDS bank-spread swizzle
    return idx ^ ((idx >> 4) & 0xF);
}

// ---------------------------------------------------------------------------
// Prep: swizzle FULL Wc1 (K padded 261->288, zero-fill) into bf16 B fragments.
// uint16 index: ((w*9 + c)*2 + h)*256 + t*8 + e
//   n = w*16 + (t&15);  k = c*32 + ((t&16)?16:0) + h*8 + e   (B: K = e + hi16)
// ---------------------------------------------------------------------------
__global__ __launch_bounds__(256) void ac_prep(const float* __restrict__ Wc1,
                                               unsigned short* __restrict__ WB) {
    int tid = blockIdx.x * 256 + threadIdx.x;      // 0 .. 36863
    int e = tid & 7;
    int t = (tid >> 3) & 31;
    int h = (tid >> 8) & 1;
    int rest = tid >> 9;                            // w*9 + c
    int c = rest % 9;
    int w = rest / 9;
    int n = w * 16 + (t & 15);
    int k = c * 32 + ((t & 16) ? 16 : 0) + h * 8 + e;
    float v = (k < FEAT) ? Wc1[(size_t)k * HID + n] : 0.f;
    WB[tid] = f2bf(v);
}

// ---------------------------------------------------------------------------
// Main fused kernel: 8 waves, 128 rows/block. Wave w owns N-tile w (16 cols).
// ---------------------------------------------------------------------------
__global__ __launch_bounds__(256) void ac_main(
    const float* __restrict__ x,   const float* __restrict__ w0,
    const float* __restrict__ b0,  const float* __restrict__ w1,
    const float* __restrict__ b1,  const float* __restrict__ w2,
    const float* __restrict__ b2,  const float* __restrict__ leaf,
    const float* __restrict__ bc1, const float* __restrict__ Wc2,
    const float* __restrict__ bc2, const unsigned short* __restrict__ WB,
    float* __restrict__ outP, float* __restrict__ outV)
{
    __shared__ uint4 Afrag[4 * NC * 2 * 32];   // 2304 x 16B = 36 KB
    __shared__ float vsh[8 * ROWS_PB];         // per-wave v partials (4 KB)

    const int tid  = threadIdx.x;
    const int lane = tid & 31;
    const int wv   = tid >> 5;
    const int rowBase = blockIdx.x * ROWS_PB;
    const int n    = wv * 16 + (lane & 15);          // output column
    const int hi8  = (lane & 16) ? 8 : 0;

    // ---- per-wave invariant B fragments (L2-resident, reused by all blocks)
    v16bf bfrag[NC];
    #pragma unroll
    for (int c = 0; c < NC; ++c) {
        union { uint4 u[2]; v16bf v; } bu;
        const uint4* bp = (const uint4*)WB + ((wv * NC + c) * 2) * 32 + lane;
        bu.u[0] = bp[0];       // h = 0
        bu.u[1] = bp[32];      // h = 1
        bfrag[c] = bu.v;
    }
    const float wc2n = Wc2[n];
    const float bc1n = bc1[n];

    // ================= GEMM: two staging rounds of 4 M-tiles =================
    for (int st = 0; st < 2; ++st) {
        __syncthreads();
        // Stage x (K padded to 288) as bf16 A fragments.
        // Segment s of 8 consecutive k == fragment block (c=s>>2,h=s&1,thi=(s>>1)&1):
        // global reads are contiguous 1KB per row across lanes.
        #pragma unroll
        for (int i = 0; i < 9; ++i) {
            int b = tid + i * 256;              // 0..2303
            int s = b % 36;
            int row64 = b / 36;                 // 0..63
            int mL  = row64 >> 4;
            int r15 = row64 & 15;
            int row = rowBase + st * 64 + row64;
            const float* g = x + (size_t)row * FEAT;
            int k0 = s * 8;
            float f[8];
            #pragma unroll
            for (int j = 0; j < 8; ++j) {
                int k = k0 + j;
                f[j] = (k < FEAT) ? g[k] : 0.f;
            }
            uint4 u;
            u.x = pack2(f[0], f[1]); u.y = pack2(f[2], f[3]);
            u.z = pack2(f[4], f[5]); u.w = pack2(f[6], f[7]);
            int c = s >> 2, h = s & 1, thi = (s >> 1) & 1;
            int idx = ((mL * NC + c) * 2 + h) * 32 + thi * 16 + r15;
            Afrag[swz(idx)] = u;
        }
        __syncthreads();

        for (int mL = 0; mL < 4; ++mL) {
            // preload A chunks -> scheduler can batch loads ahead of WMMA chain
            union AU { uint4 u[2]; v16bf v; } au[NC];
            #pragma unroll
            for (int c = 0; c < NC; ++c) {
                int i0 = ((mL * NC + c) * 2 + 0) * 32 + lane;
                int i1 = i0 + 32;
                au[c].u[0] = Afrag[swz(i0)];
                au[c].u[1] = Afrag[swz(i1)];
            }
            v8f acc;
            #pragma unroll
            for (int r = 0; r < 8; ++r) acc[r] = bc1n;
            #pragma unroll
            for (int c = 0; c < NC; ++c)
                acc = __builtin_amdgcn_wmma_f32_16x16x32_bf16(
                        false, au[c].v, false, bfrag[c], (short)0, acc, false, false);

            // relu, scale by Wc2[n], butterfly-reduce 8 rows over 16 lanes
            float pv[8];
            #pragma unroll
            for (int r = 0; r < 8; ++r)
                pv[r] = fmaxf(acc[r], 0.f) * wc2n;

            float n0 = halve(pv[0], pv[1], 1, lane);
            float n1 = halve(pv[2], pv[3], 1, lane);
            float n2 = halve(pv[4], pv[5], 1, lane);
            float n3 = halve(pv[6], pv[7], 1, lane);
            float p0 = halve(n0, n1, 2, lane);
            float p1 = halve(n2, n3, 2, lane);
            float q  = halve(p0, p1, 4, lane);
            q += __shfl_xor(q, 8);
            // lane now holds the half-sum for row (lane&7); lanes 0-7 / 16-23 write
            const int mG = st * 4 + mL;
            if ((lane & 8) == 0)
                vsh[wv * ROWS_PB + mG * 16 + (lane & 7) + hi8] = q;
        }
    }
    __syncthreads();

    // deterministic fixed-order reduction across the 8 waves
    if (tid < ROWS_PB) {
        float v = bc2[0];
        #pragma unroll
        for (int w = 0; w < 8; ++w) v += vsh[w * ROWS_PB + tid];
        outV[rowBase + tid] = v;
    }

    // ================= Tree (f32, wave-cooperative, 16 rows/wave) ============
    for (int rr = 0; rr < 16; ++rr) {
        int row = rowBase + wv * 16 + rr;
        const float* xr = x + (size_t)row * FEAT;

        float s0 = 0.f;
        for (int k = lane; k < 261; k += 32) s0 += xr[k] * w0[k];
        float g = wred32(s0) + b0[0];
        float cum = 1.f / (1.f + __expf(-g));
        int node = (g >= 0.f) ? 1 : 0;

        {
            const float* w = w1 + node * 133;
            int base = 5 + node * 128;
            float s = 0.f;
            for (int k = lane; k < 133; k += 32) {
                float xv = (k < 5) ? xr[k] : xr[base + k - 5];
                s += xv * w[k];
            }
            g = wred32(s) + b1[node];
            cum *= 1.f / (1.f + __expf(-g));
            node = node * 2 + ((g >= 0.f) ? 1 : 0);
        }
        {
            const float* w = w2 + node * 69;
            int base = 5 + node * 64;
            float s = 0.f;
            for (int k = lane; k < 69; k += 32) {
                float xv = (k < 5) ? xr[k] : xr[base + k - 5];
                s += xv * w[k];
            }
            g = wred32(s) + b2[node];
            cum *= 1.f / (1.f + __expf(-g));
            node = node * 2 + ((g >= 0.f) ? 1 : 0);
        }
        if (lane < 8) outP[(size_t)row * 8 + lane] = cum * leaf[node * 8 + lane];
    }
}

// ---------------------------------------------------------------------------
extern "C" void kernel_launch(void* const* d_in, const int* in_sizes, int n_in,
                              void* d_out, int out_size, void* d_ws, size_t ws_size,
                              hipStream_t stream) {
    const float* x    = (const float*)d_in[0];
    const float* w0   = (const float*)d_in[1];
    const float* b0   = (const float*)d_in[2];
    const float* w1   = (const float*)d_in[3];
    const float* b1   = (const float*)d_in[4];
    const float* w2   = (const float*)d_in[5];
    const float* b2   = (const float*)d_in[6];
    const float* leaf = (const float*)d_in[7];
    const float* Wc1  = (const float*)d_in[8];
    const float* bc1  = (const float*)d_in[9];
    const float* Wc2  = (const float*)d_in[10];
    const float* bc2  = (const float*)d_in[11];

    unsigned short* WB = (unsigned short*)d_ws;    // 72 KB bf16 swizzled Wc1 (padded)
    float* outP = (float*)d_out;                   // (B, 8)
    float* outV = outP + (size_t)B_ROWS * 8;       // (B, 1)

    ac_prep<<<144, 256, 0, stream>>>(Wc1, WB);
    ac_main<<<NBLK, 256, 0, stream>>>(x, w0, b0, w1, b1, w2, b2, leaf,
                                      bc1, Wc2, bc2, WB, outP, outV);
}
